// NeuralCDE_46651934769910
// MI455X (gfx1250) — matrix operand order repaired and checked by reading, not compiled, not run on hardware
//
#include <hip/hip_runtime.h>

// Neural CDE RK4 scan for MI455X (gfx1250, wave32, WMMA).
// One persistent workgroup per 16-row batch tile; weights live in registers
// as WMMA B-fragments; activations stage through LDS.

typedef _Float16 f16;
typedef __attribute__((ext_vector_type(16))) _Float16 v16h;
typedef __attribute__((ext_vector_type(8)))  _Float16 v8h;
typedef __attribute__((ext_vector_type(8)))  float    v8f;

#define BATCH 512
#define SEQ   500
#define IC    8      // input channels
#define HID   64     // hidden
#define HH    128    // hidden_hidden
#define MT    16     // batch rows per workgroup

// ---- fast math --------------------------------------------------------------

__device__ __forceinline__ float fast_rcp(float x) {
#if __has_builtin(__builtin_amdgcn_rcpf)
    return __builtin_amdgcn_rcpf(x);
#else
    return 1.0f / x;
#endif
}

__device__ __forceinline__ float fast_tanh(float x) {
#if __has_builtin(__builtin_amdgcn_tanhf)
    return __builtin_amdgcn_tanhf(x);          // native v_tanh_f32 (TRANS32)
#elif __has_builtin(__builtin_amdgcn_tanh_f32)
    return __builtin_amdgcn_tanh_f32(x);
#else
    const float e = __expf(2.0f * x);          // v_exp_f32 based fallback
    return 1.0f - 2.0f * fast_rcp(e + 1.0f);
#endif
}

// ---- WMMA helpers -----------------------------------------------------------

__device__ __forceinline__ v8f wmma_f16(v16h a, v16h b, v8f c) {
    // D = A(16x32 f16) * B(32x16 f16) + C(16x16 f32)
    return __builtin_amdgcn_wmma_f32_16x16x32_f16(
        /*neg_a=*/false, a, /*neg_b=*/false, b,
        /*c_mod=*/(short)0, c, /*reuse_a=*/false, /*reuse_b=*/false);
}

// A-fragment (16-bit A 16x32 layout, ISA 7.12.2):
// lane = {half = lane>>4, m = lane&15}; elements 0..7  = A[m][kc + half*8 .. +7]
//                                       elements 8..15 = A[m][kc + 16 + half*8 .. +7]
__device__ __forceinline__ v16h load_a_lds(const f16* base, int ldk, int m, int half, int kc) {
    const f16* p = base + m * ldk + kc + half * 8;
    v16h a;
    *((v8h*)&a)     = *(const v8h*)(p);
    *((v8h*)&a + 1) = *(const v8h*)(p + 16);
    return a;
}

// B-fragment (32x16): lane L holds column N=L&15; elements 0..7 = K 8*(L>>4)+e, 8..15 = K 16+8*(L>>4)+(e-8).
// Built once from global f32 weights (row-major KxN_total), converted to f16.
__device__ __forceinline__ v16h load_b_gl(const float* __restrict__ W, int ldn,
                                          int krow, int ncol) {
    const int l = krow & 31;
    const int kbase = krow - l;
    const int hh = l >> 4;
    const float* p = W + (size_t)kbase * ldn + ncol + (l & 15);
    v16h b;
#pragma unroll
    for (int e = 0; e < 16; ++e) b[e] = (f16)p[(size_t)((e < 8) ? (8 * hh + e) : (16 + 8 * hh + e - 8)) * ldn];
    return b;
}

// ---- Kernel -----------------------------------------------------------------

__global__ __launch_bounds__(256) void neural_cde_rk4(
    const float* __restrict__ times,
    const float* __restrict__ ca,   // coeff_a       (B, SEQ-1, IC)
    const float* __restrict__ cb,   // coeff_b
    const float* __restrict__ c2,   // coeff_two_c
    const float* __restrict__ c3,   // coeff_three_d
    const float* __restrict__ iW,   // init_W (IC, HID)
    const float* __restrict__ ib,   // init_b (HID)
    const float* __restrict__ W1,   // f_W1 (HID, HH)
    const float* __restrict__ b1,   // f_b1 (HH)
    const float* __restrict__ W2,   // f_W2 (HH, HID*IC)
    const float* __restrict__ b2,   // f_b2 (HID*IC)
    const float* __restrict__ lW,   // lin_W (HID, 1)
    const float* __restrict__ lb,   // lin_b (1)
    float* __restrict__ out)        // (B, 1)
{
    __shared__ float sZ[MT * HID];        // z, f32 master copy
    __shared__ float sAcc[MT * HID];      // RK4 weighted-k accumulator
    __shared__ f16   sZA[MT * HID];       // stage input (A of GEMM1), f16
    __shared__ f16   sH[MT * HH];         // relu(h), A of GEMM2, f16
    __shared__ float sV[MT * HID * IC];   // tanh output, f32
    __shared__ float sCb[MT * IC], sC2[MT * IC], sC3[MT * IC];
    __shared__ float sh_h;

    const int tid  = threadIdx.x;
    const int wave = tid >> 5;
    const int lane = tid & 31;
    const int m16  = lane & 15;
    const int half = lane >> 4;
    const int b0   = blockIdx.x * MT;

    // ---- persistent register-resident weight fragments ----
    v16h w1b0 = load_b_gl(W1, HH, 0 + lane, wave * 16);   // K chunk 0..31
    v16h w1b1 = load_b_gl(W1, HH, 32 + lane, wave * 16);  // K chunk 32..63
    float b1v = b1[wave * 16 + m16];
    v16h  w2b[4][4];
    float b2v[4];
#pragma unroll
    for (int nt = 0; nt < 4; ++nt) {
        const int nc = (wave * 4 + nt) * 16;
#pragma unroll
        for (int c = 0; c < 4; ++c)
            w2b[nt][c] = load_b_gl(W2, HID * IC, c * 32 + lane, nc);
        b2v[nt] = b2[nc + m16];
    }

    // ---- z0 = X(t0) @ init_W + init_b ;  X(t0) = coeff_a[:,0,:] ----
    {
        const int m  = tid >> 4;
        const int h0 = (tid & 15) * 4;
        float zz[4];
#pragma unroll
        for (int j = 0; j < 4; ++j) zz[j] = ib[h0 + j];
        const float* car = ca + (size_t)(b0 + m) * (SEQ - 1) * IC;
#pragma unroll
        for (int c = 0; c < IC; ++c) {
            const float x = car[c];
#pragma unroll
            for (int j = 0; j < 4; ++j) zz[j] += x * iW[c * HID + h0 + j];
        }
#pragma unroll
        for (int j = 0; j < 4; ++j) {
            sZ[m * HID + h0 + j]  = zz[j];
            sZA[m * HID + h0 + j] = (f16)zz[j];
        }
    }

    // ---- time scan ----
    for (int i = 0; i < SEQ - 1; ++i) {
        __syncthreads();  // prior-step LDS reads done before overwriting step data
        if (tid == 0) sh_h = times[i + 1] - times[i];
        if (tid < MT * IC) {
            const int m = tid >> 3, c = tid & 7;
            const size_t off = ((size_t)(b0 + m) * (SEQ - 1) + i) * IC + c;
            sCb[tid] = cb[off]; sC2[tid] = c2[off]; sC3[tid] = c3[off];
        }

#pragma unroll
        for (int s = 0; s < 4; ++s) {
            __syncthreads();  // guards sZA/sAcc (prev contraction) + step loads
            const float h = sh_h;

            // GEMM1: h = relu(z @ W1 + b1)   (16x64)@(64x128), 1 N-tile per wave
            {
                v16h a0 = load_a_lds(sZA, HID, m16, half, 0);
                v16h a1 = load_a_lds(sZA, HID, m16, half, 32);
                v8f  cc = {};
                cc = wmma_f16(a0, w1b0, cc);
                cc = wmma_f16(a1, w1b1, cc);
#pragma unroll
                for (int r = 0; r < 8; ++r) {
                    float hv = cc[r] + b1v;
                    hv = hv > 0.0f ? hv : 0.0f;
                    // D layout: m = r + 8*half, n = m16
                    sH[(r + 8 * half) * HH + wave * 16 + m16] = (f16)hv;
                }
            }
            __syncthreads();

            // GEMM2: v = tanh(h @ W2 + b2)   (16x128)@(128x512), 4 N-tiles per wave
            {
                v16h a[4];
#pragma unroll
                for (int c = 0; c < 4; ++c) a[c] = load_a_lds(sH, HH, m16, half, c * 32);
#pragma unroll
                for (int nt = 0; nt < 4; ++nt) {
                    v8f cc = {};
#pragma unroll
                    for (int c = 0; c < 4; ++c) cc = wmma_f16(a[c], w2b[nt][c], cc);
                    const int ncol = (wave * 4 + nt) * 16 + m16;
#pragma unroll
                    for (int r = 0; r < 8; ++r) {
                        const float v = fast_tanh(cc[r] + b2v[nt]);
                        sV[(r + 8 * half) * (HID * IC) + ncol] = v;
                    }
                }
            }
            __syncthreads();

            // dX/dt + k = einsum('hi,i->h', v.reshape(HID,IC), dx) ; RK4 update
            // All stages of step i share spline piece idx = i; frac = {0,h/2,h/2,h}.
            {
                const int m  = tid >> 4;
                const int h0 = (tid & 15) * 4;
                float dxv[IC];
#pragma unroll
                for (int c = 0; c < IC; ++c) {
                    if (s == 0) {
                        dxv[c] = sCb[m * IC + c];
                    } else {
                        const float frac = (s == 3) ? h : 0.5f * h;
                        dxv[c] = sCb[m * IC + c] +
                                 (sC2[m * IC + c] + sC3[m * IC + c] * frac) * frac;
                    }
                }
                const float wk = (s == 0 || s == 3) ? h * (1.0f / 6.0f) : h * (1.0f / 3.0f);
                const float an = (s == 2) ? h : 0.5f * h;  // coeff for next stage input
#pragma unroll
                for (int j = 0; j < 4; ++j) {
                    const int hid = h0 + j;
                    const float* vp = sV + m * (HID * IC) + hid * IC;
                    float k = 0.0f;
#pragma unroll
                    for (int c = 0; c < IC; ++c) k += vp[c] * dxv[c];
                    const float zc   = sZ[m * HID + hid];
                    const float accv = (s == 0) ? wk * k : sAcc[m * HID + hid] + wk * k;
                    sAcc[m * HID + hid] = accv;
                    if (s < 3) {
                        sZA[m * HID + hid] = (f16)(zc + an * k);
                    } else {
                        const float zn = zc + accv;   // z + (h/6)(k1+2k2+2k3+k4)
                        sZ[m * HID + hid]  = zn;
                        sZA[m * HID + hid] = (f16)zn;
                    }
                }
            }
        }
    }

    // ---- readout: sigmoid(z @ lin_W + lin_b) ----
    __syncthreads();
    if (tid < MT) {
        float acc = lb[0];
#pragma unroll
        for (int hid = 0; hid < HID; ++hid) acc += sZ[tid * HID + hid] * lW[hid];
        out[b0 + tid] = fast_rcp(1.0f + __expf(-acc));
    }
}

// ---- launch -----------------------------------------------------------------

extern "C" void kernel_launch(void* const* d_in, const int* in_sizes, int n_in,
                              void* d_out, int out_size, void* d_ws, size_t ws_size,
                              hipStream_t stream) {
    const float* times = (const float*)d_in[0];
    const float* ca    = (const float*)d_in[1];
    const float* cb    = (const float*)d_in[2];
    const float* c2    = (const float*)d_in[3];
    const float* c3    = (const float*)d_in[4];
    const float* iW    = (const float*)d_in[5];
    const float* ib    = (const float*)d_in[6];
    const float* W1    = (const float*)d_in[7];
    const float* b1    = (const float*)d_in[8];
    const float* W2    = (const float*)d_in[9];
    const float* b2    = (const float*)d_in[10];
    const float* lW    = (const float*)d_in[11];
    const float* lb    = (const float*)d_in[12];
    float* out = (float*)d_out;

    dim3 grid(BATCH / MT);   // 32 workgroups
    dim3 block(256);         // 8 wave32 waves
    neural_cde_rk4<<<grid, block, 0, stream>>>(times, ca, cb, c2, c3,
                                               iW, ib, W1, b1, W2, b2, lW, lb, out);
}
